// TransformerEncoder_13649406067499
// MI455X (gfx1250) — matrix operand order repaired and checked by reading, not compiled, not run on hardware
//
#include <hip/hip_runtime.h>
#include <hip/hip_bf16.h>
#include <math.h>

// Problem constants (match reference)
#define D_    1024
#define H_    16
#define DK_   64
#define NL_   4
#define DFF_  4096
#define ROT_  32
#define B_    2
#define L_    1024
#define M_    (B_*L_)        // 2048 rows of activations
#define HD_   (H_*DK_)       // 1024

typedef __attribute__((ext_vector_type(16))) __bf16 v16bf;
typedef __attribute__((ext_vector_type(8)))  float  v8f;

static __device__ __forceinline__ v8f wmma_bf16(v16bf a, v16bf b, v8f c) {
  // D(f32,16x16) = A(bf16,16x32) * B(bf16,32x16) + C
  return __builtin_amdgcn_wmma_f32_16x16x32_bf16(false, a, false, b, (short)0, c,
                                                 false, false);
}

// A-fragment K index (contiguous order, matches B frags):
// lanes0-15: K=0..15, lanes16-31: K=16..31 (element j -> K = 16*half + j)
static __device__ __forceinline__ int afrag_k(int j, int half) {
  return half * 16 + j;
}

#define TSTR 34   // padded LDS stride (halves) -> conflict-free frag gathers
#define KSTR 66   // padded stride for 64-wide K rows in attention

// ---------------------------------------------------------------------------
// Weight pre-pass: Wt[n][k] = (bf16) W[k][n]   (transpose + fp32->bf16)
// 32x32 tiles through LDS; weights then live bf16 [N][K] so GEMM B-frags are
// 16 contiguous halves (one global_load_b128 pair per fragment).
// ---------------------------------------------------------------------------
__global__ __launch_bounds__(256)
void wt_convert_kernel(const float* __restrict__ W, __bf16* __restrict__ Wt,
                       int Kdim, int Ndim)
{
  __shared__ float tile[32][33];
  const int k0 = blockIdx.y * 32;
  const int n0 = blockIdx.x * 32;
  const int tx = threadIdx.x & 31;
  const int ty = threadIdx.x >> 5;       // 0..7
  #pragma unroll
  for (int i = 0; i < 4; ++i) {
    const int k = ty + i * 8;
    tile[k][tx] = W[(size_t)(k0 + k) * Ndim + n0 + tx];
  }
  __syncthreads();
  #pragma unroll
  for (int i = 0; i < 4; ++i) {
    const int n = ty + i * 8;
    Wt[(size_t)(n0 + n) * Kdim + k0 + tx] = (__bf16)tile[tx][n];
  }
}

// ---------------------------------------------------------------------------
// LayerNorm: out = LN(inA*scaleA + inB) * g + beta   (inB optional)
// one 256-thread block per row of D_=1024
// ---------------------------------------------------------------------------
__global__ __launch_bounds__(256)
void ln_kernel(const float* __restrict__ inA, const float* __restrict__ inB,
               const float* __restrict__ g, const float* __restrict__ beta,
               float* __restrict__ out, float scaleA)
{
  __shared__ float red[256];
  const int row = blockIdx.x;
  const int tid = threadIdx.x;
  const size_t base = (size_t)row * D_ + tid * 4;

  float4 v = *reinterpret_cast<const float4*>(inA + base);
  v.x *= scaleA; v.y *= scaleA; v.z *= scaleA; v.w *= scaleA;
  if (inB) {
    const float4 r = *reinterpret_cast<const float4*>(inB + base);
    v.x += r.x; v.y += r.y; v.z += r.z; v.w += r.w;
  }

  red[tid] = v.x + v.y + v.z + v.w;
  __syncthreads();
  for (int s = 128; s >= 1; s >>= 1) {
    if (tid < s) red[tid] += red[tid + s];
    __syncthreads();
  }
  const float mu = red[0] * (1.0f / D_);
  __syncthreads();

  const float dx = v.x - mu, dy = v.y - mu, dz = v.z - mu, dw = v.w - mu;
  red[tid] = dx*dx + dy*dy + dz*dz + dw*dw;
  __syncthreads();
  for (int s = 128; s >= 1; s >>= 1) {
    if (tid < s) red[tid] += red[tid + s];
    __syncthreads();
  }
  const float rs = rsqrtf(red[0] * (1.0f / D_) + 1e-6f);

  const float4 gg = *reinterpret_cast<const float4*>(g + tid * 4);
  const float4 bb = *reinterpret_cast<const float4*>(beta + tid * 4);
  float4 o;
  o.x = dx * rs * gg.x + bb.x;
  o.y = dy * rs * gg.y + bb.y;
  o.z = dz * rs * gg.z + bb.z;
  o.w = dw * rs * gg.w + bb.w;
  *reinterpret_cast<float4*>(out + base) = o;
}

// ---------------------------------------------------------------------------
// GEMM: C[M,N] = act(A[M,K] @ Wt^T + bias), Wt is bf16 [N][K] (pre-transposed)
// block = 256 threads (8 wave32), tile 128x128, K step 32.
// A staged through double-buffered LDS (fp32->bf16); B frags loaded straight
// from global bf16 (L2-resident weights). One barrier per K step.
// ---------------------------------------------------------------------------
__global__ __launch_bounds__(256)
void gemm_bf16_kernel(const float* __restrict__ A, const __bf16* __restrict__ Wt,
                      const float* __restrict__ bias, float* __restrict__ C,
                      int Mdim, int Ndim, int Kdim, int relu)
{
  __shared__ __bf16 As[2][128 * TSTR];   // [row][k] row-major, stride 34

  const int tid  = threadIdx.x;
  const int lane = tid & 31;
  const int wid  = tid >> 5;
  const int half = lane >> 4;
  const int l15  = lane & 15;
  const int m0 = blockIdx.y * 128;
  const int n0 = blockIdx.x * 128;
  const int waveM = wid >> 1;   // 0..3 -> 32-row strip
  const int waveN = wid & 1;    // 0..1 -> 64-col strip

  // staging addresses for this thread (4 float4 per tile)
  const int arow = tid >> 3;           // base row (stepped by 32 over 4 passes)
  const int akc  = (tid & 7) * 4;

  v8f acc[2][4];
  #pragma unroll
  for (int i = 0; i < 2; ++i)
    #pragma unroll
    for (int j = 0; j < 4; ++j)
      #pragma unroll
      for (int e = 0; e < 8; ++e) acc[i][j][e] = 0.0f;

  float4 rg[4];
  // ---- prologue: stage tile 0 into buffer 0 ----
  #pragma unroll
  for (int p = 0; p < 4; ++p)
    rg[p] = *reinterpret_cast<const float4*>(
        A + (size_t)(m0 + arow + p * 32) * Kdim + akc);
  #pragma unroll
  for (int p = 0; p < 4; ++p) {
    __bf16* d = &As[0][(arow + p * 32) * TSTR + akc];
    d[0] = (__bf16)rg[p].x; d[1] = (__bf16)rg[p].y;
    d[2] = (__bf16)rg[p].z; d[3] = (__bf16)rg[p].w;
  }
  __syncthreads();

  for (int k0 = 0; k0 < Kdim; k0 += 32) {
    const int  cur  = (k0 >> 5) & 1;
    const bool more = (k0 + 32) < Kdim;

    // issue next A tile's global loads early (overlap with WMMA below)
    if (more) {
      #pragma unroll
      for (int p = 0; p < 4; ++p)
        rg[p] = *reinterpret_cast<const float4*>(
            A + (size_t)(m0 + arow + p * 32) * Kdim + k0 + 32 + akc);
    }

    // B fragments straight from global bf16 [N][K]: 16 contiguous halves
    v16bf bfr[4];
    #pragma unroll
    for (int nt = 0; nt < 4; ++nt) {
      const int col = n0 + waveN * 64 + nt * 16 + l15;
      bfr[nt] = *reinterpret_cast<const v16bf*>(
          Wt + (size_t)col * Kdim + k0 + half * 16);
    }
    // A fragments from LDS (conflict-free padded gathers)
    v16bf af[2];
    #pragma unroll
    for (int mt = 0; mt < 2; ++mt) {
      const __bf16* p = &As[cur][(waveM * 32 + mt * 16 + l15) * TSTR];
      #pragma unroll
      for (int j = 0; j < 16; ++j) af[mt][j] = p[afrag_k(j, half)];
    }
    #pragma unroll
    for (int mt = 0; mt < 2; ++mt)
      #pragma unroll
      for (int nt = 0; nt < 4; ++nt)
        acc[mt][nt] = wmma_bf16(af[mt], bfr[nt], acc[mt][nt]);

    // convert + park next tile in the other buffer
    if (more) {
      #pragma unroll
      for (int p = 0; p < 4; ++p) {
        __bf16* d = &As[cur ^ 1][(arow + p * 32) * TSTR + akc];
        d[0] = (__bf16)rg[p].x; d[1] = (__bf16)rg[p].y;
        d[2] = (__bf16)rg[p].z; d[3] = (__bf16)rg[p].w;
      }
    }
    __syncthreads();
  }

  // ---- epilogue: bias / relu / store (f32 C layout: row = r + 8*half) ----
  #pragma unroll
  for (int nt = 0; nt < 4; ++nt) {
    const int col = n0 + waveN * 64 + nt * 16 + l15;
    const float bv = bias ? bias[col] : 0.0f;
    #pragma unroll
    for (int mt = 0; mt < 2; ++mt) {
      #pragma unroll
      for (int r = 0; r < 8; ++r) {
        const int row = m0 + waveM * 32 + mt * 16 + r + 8 * half;
        float v = acc[mt][nt][r] + bv;
        if (relu) v = fmaxf(v, 0.0f);
        C[(size_t)row * Ndim + col] = v;
      }
    }
  }
}

// ---------------------------------------------------------------------------
// RoPE (reference quirk: tables use freqs[-H:], so the angle depends on the
// HEAD index, not the sequence position): angle = (L-H+h) * BASE^(-j/16)
// ---------------------------------------------------------------------------
__global__ __launch_bounds__(256)
void rope_kernel(float* __restrict__ t, int seqlen)
{
  const int i = blockIdx.x * blockDim.x + threadIdx.x;
  const int pair = i & 15;
  const int h    = (i >> 4) & (H_ - 1);
  const int bl   = i >> 8;                 // b*L + l
  if (bl >= B_ * L_) return;
  const size_t base = ((size_t)bl * H_ + h) * DK_;

  const float pos  = (float)(seqlen - H_ + h);
  const float invf = __expf(-((float)pair / 16.0f) * 8.317766166719343f); // ln(4096)
  const float ang  = pos * invf;
  const float c = __cosf(ang), s = __sinf(ang);

  const float a0 = t[base + pair];
  const float a1 = t[base + pair + 16];
  t[base + pair]      = a0 * c - a1 * s;
  t[base + pair + 16] = a1 * c + a0 * s;
}

// ---------------------------------------------------------------------------
// Flash attention: softmax(Q K^T / sqrt(DK)) V, per (b,h), online softmax.
// grid = (L/128, B*H), block = 256 (8 waves); each wave owns 16 query rows.
// K and V tiles staged cooperatively into bf16 LDS (V transposed), so frag
// gathers are contiguous ds pairs instead of 8x-redundant global gathers.
// ---------------------------------------------------------------------------
__global__ __launch_bounds__(256)
void attn_kernel(const float* __restrict__ Q, const float* __restrict__ Km,
                 const float* __restrict__ V, float* __restrict__ O)
{
  __shared__ __bf16 Ks[32 * KSTR];          // [key][dk]  stride 66
  __shared__ __bf16 Vs[64 * TSTR];          // [dk][key]  stride 34 (transposed)
  __shared__ __bf16 pbuf[8 * 16 * TSTR];    // per-wave P staging (C->A relayout)

  const int tid  = threadIdx.x;
  const int lane = tid & 31;
  const int wid  = tid >> 5;
  const int half = lane >> 4;
  const int l15  = lane & 15;

  const int bh = blockIdx.y;
  const int b  = bh / H_;
  const int h  = bh % H_;
  const int qrow0 = blockIdx.x * 128 + wid * 16;
  const size_t base = ((size_t)b * L_ * H_ + h) * DK_;   // (b, l=0, h, 0)
  const float inv_temp = 0.125f;                          // 1/sqrt(64)

  // load + scale Q fragments (A layout), 2 chunks of K-dim 32 covering DK=64
  v16bf qf[2];
  {
    const float* qp = Q + base + (size_t)(qrow0 + l15) * HD_;
    #pragma unroll
    for (int ch = 0; ch < 2; ++ch)
      #pragma unroll
      for (int j = 0; j < 16; ++j)
        qf[ch][j] = (__bf16)(qp[ch * 32 + afrag_k(j, half)] * inv_temp);
  }

  v8f o[4];
  float mrow[8], lrow[8];
  #pragma unroll
  for (int t = 0; t < 4; ++t)
    #pragma unroll
    for (int e = 0; e < 8; ++e) o[t][e] = 0.0f;
  #pragma unroll
  for (int r = 0; r < 8; ++r) { mrow[r] = -1e30f; lrow[r] = 0.0f; }

  __bf16* pw = &pbuf[wid * 16 * TSTR];
  const int skey = tid >> 4;            // staging: key row 0..15 (+16 on pass 1)
  const int sc4  = (tid & 15) * 4;      // staging: dk column group

  for (int j0 = 0; j0 < L_; j0 += 32) {
    // ---- cooperative staging of K and V tiles (32 keys x 64 dims) ----
    #pragma unroll
    for (int p = 0; p < 2; ++p) {
      const int key = skey + p * 16;
      const float4 k4 = *reinterpret_cast<const float4*>(
          Km + base + (size_t)(j0 + key) * HD_ + sc4);
      const float4 v4 = *reinterpret_cast<const float4*>(
          V  + base + (size_t)(j0 + key) * HD_ + sc4);
      __bf16* kd = &Ks[key * KSTR + sc4];
      kd[0] = (__bf16)k4.x; kd[1] = (__bf16)k4.y;
      kd[2] = (__bf16)k4.z; kd[3] = (__bf16)k4.w;
      Vs[(sc4 + 0) * TSTR + key] = (__bf16)v4.x;
      Vs[(sc4 + 1) * TSTR + key] = (__bf16)v4.y;
      Vs[(sc4 + 2) * TSTR + key] = (__bf16)v4.z;
      Vs[(sc4 + 3) * TSTR + key] = (__bf16)v4.w;
    }
    __syncthreads();

    // ---- S = Q K^T for 32 keys -> two 16x16 f32 tiles ----
    v8f s[2];
    #pragma unroll
    for (int nt = 0; nt < 2; ++nt) {
      #pragma unroll
      for (int e = 0; e < 8; ++e) s[nt][e] = 0.0f;
      #pragma unroll
      for (int ch = 0; ch < 2; ++ch) {
        v16bf kf;   // B layout: col = key, k = dk (halves split K range)
        const __bf16* kp = &Ks[(nt * 16 + l15) * KSTR + ch * 32 + half * 16];
        #pragma unroll
        for (int j = 0; j < 16; ++j) kf[j] = kp[j];
        s[nt] = wmma_bf16(qf[ch], kf, s[nt]);
      }
    }

    // ---- online softmax per query row (row r+8*half lives in a 16-lane half)
    #pragma unroll
    for (int r = 0; r < 8; ++r) {
      float vmax = fmaxf(s[0][r], s[1][r]);
      #pragma unroll
      for (int off = 8; off >= 1; off >>= 1)
        vmax = fmaxf(vmax, __shfl_xor(vmax, off, 32));
      const float mnew = fmaxf(mrow[r], vmax);
      const float p0 = __expf(s[0][r] - mnew);
      const float p1 = __expf(s[1][r] - mnew);
      float rsum = p0 + p1;
      #pragma unroll
      for (int off = 8; off >= 1; off >>= 1)
        rsum += __shfl_xor(rsum, off, 32);
      const float scale = __expf(mrow[r] - mnew);
      lrow[r] = lrow[r] * scale + rsum;
      mrow[r] = mnew;
      #pragma unroll
      for (int t = 0; t < 4; ++t) o[t][r] *= scale;
      pw[(r + 8 * half) * TSTR + l15]      = (__bf16)p0;
      pw[(r + 8 * half) * TSTR + 16 + l15] = (__bf16)p1;
    }

    // ---- P (16x32) as A fragment, then O += P V ----
    v16bf pf;
    {
      const __bf16* pp = &pw[l15 * TSTR];
      #pragma unroll
      for (int j = 0; j < 16; ++j) pf[j] = pp[afrag_k(j, half)];
    }
    #pragma unroll
    for (int t = 0; t < 4; ++t) {
      v16bf vf;   // B layout: col = dk, k = key
      const __bf16* vp = &Vs[(t * 16 + l15) * TSTR + half * 16];
      #pragma unroll
      for (int j = 0; j < 16; ++j) vf[j] = vp[j];
      o[t] = wmma_bf16(pf, vf, o[t]);
    }
    __syncthreads();   // protect Ks/Vs before next staging pass
  }

  // ---- normalize and store ----
  #pragma unroll
  for (int t = 0; t < 4; ++t)
    #pragma unroll
    for (int r = 0; r < 8; ++r) {
      const int row = qrow0 + r + 8 * half;
      O[base + (size_t)row * HD_ + t * 16 + l15] = o[t][r] / lrow[r];
    }
}

// ---------------------------------------------------------------------------
extern "C" void kernel_launch(void* const* d_in, const int* in_sizes, int n_in,
                              void* d_out, int out_size, void* d_ws, size_t ws_size,
                              hipStream_t stream) {
  (void)in_sizes; (void)n_in; (void)out_size; (void)ws_size;
  const float* src   = (const float*)d_in[0];
  const float* Wq    = (const float*)d_in[1];
  const float* Wk    = (const float*)d_in[2];
  const float* Wv    = (const float*)d_in[3];
  const float* Wo    = (const float*)d_in[4];
  const float* ln1_g = (const float*)d_in[5];
  const float* ln1_b = (const float*)d_in[6];
  const float* W1    = (const float*)d_in[7];
  const float* b1    = (const float*)d_in[8];
  const float* W2    = (const float*)d_in[9];
  const float* b2    = (const float*)d_in[10];
  const float* ln2_g = (const float*)d_in[11];
  const float* ln2_b = (const float*)d_in[12];
  const float* ln0_g = (const float*)d_in[13];
  const float* ln0_b = (const float*)d_in[14];
  float* out = (float*)d_out;

  // workspace: 20M floats (80MB) + 12M bf16 (24MB) transposed weights
  float* x    = (float*)d_ws;
  float* q    = x    + (size_t)M_ * D_;
  float* k    = q    + (size_t)M_ * HD_;
  float* v    = k    + (size_t)M_ * HD_;
  float* ao   = v    + (size_t)M_ * HD_;
  float* tmp  = ao   + (size_t)M_ * HD_;
  float* hbuf = tmp  + (size_t)M_ * D_;
  __bf16* wtq = (__bf16*)(hbuf + (size_t)M_ * DFF_);
  __bf16* wtk = wtq + (size_t)D_ * HD_;
  __bf16* wtv = wtk + (size_t)D_ * HD_;
  __bf16* wto = wtv + (size_t)D_ * HD_;
  __bf16* wt1 = wto + (size_t)HD_ * D_;
  __bf16* wt2 = wt1 + (size_t)D_ * DFF_;

  const dim3 blk(256);
  const dim3 gN1(D_   / 128, M_ / 128);   // (8,16)  N=1024
  const dim3 gN4(DFF_ / 128, M_ / 128);   // (32,16) N=4096
  const dim3 gAt(L_ / 128, B_ * H_);      // (8,32)
  const dim3 gT11(D_ / 32,   D_ / 32);    // transpose 1024x1024
  const dim3 gT14(DFF_ / 32, D_ / 32);    // transpose K=1024 -> N=4096
  const dim3 gT41(D_ / 32,   DFF_ / 32);  // transpose K=4096 -> N=1024

  // x = LN0(src * sqrt(D))
  ln_kernel<<<M_, blk, 0, stream>>>(src, nullptr, ln0_g, ln0_b, x, 32.0f);

  for (int layer = 0; layer < NL_; ++layer) {
    const float* wq  = Wq + (size_t)layer * D_ * HD_;
    const float* wk  = Wk + (size_t)layer * D_ * HD_;
    const float* wv  = Wv + (size_t)layer * D_ * HD_;
    const float* wo  = Wo + (size_t)layer * HD_ * D_;
    const float* w1  = W1 + (size_t)layer * D_ * DFF_;
    const float* c1  = b1 + (size_t)layer * DFF_;
    const float* w2  = W2 + (size_t)layer * DFF_ * D_;
    const float* c2  = b2 + (size_t)layer * D_;
    const float* g1  = ln1_g + (size_t)layer * D_;
    const float* be1 = ln1_b + (size_t)layer * D_;
    const float* g2  = ln2_g + (size_t)layer * D_;
    const float* be2 = ln2_b + (size_t)layer * D_;

    // one-time per-layer weight bf16 transpose (L2-resident afterwards)
    wt_convert_kernel<<<gT11, blk, 0, stream>>>(wq, wtq, D_,  HD_);
    wt_convert_kernel<<<gT11, blk, 0, stream>>>(wk, wtk, D_,  HD_);
    wt_convert_kernel<<<gT11, blk, 0, stream>>>(wv, wtv, D_,  HD_);
    wt_convert_kernel<<<gT11, blk, 0, stream>>>(wo, wto, HD_, D_);
    wt_convert_kernel<<<gT14, blk, 0, stream>>>(w1, wt1, D_,  DFF_);
    wt_convert_kernel<<<gT41, blk, 0, stream>>>(w2, wt2, DFF_, D_);

    gemm_bf16_kernel<<<gN1, blk, 0, stream>>>(x, wtq, nullptr, q, M_, HD_, D_, 0);
    gemm_bf16_kernel<<<gN1, blk, 0, stream>>>(x, wtk, nullptr, k, M_, HD_, D_, 0);
    gemm_bf16_kernel<<<gN1, blk, 0, stream>>>(x, wtv, nullptr, v, M_, HD_, D_, 0);

    const int nrope = B_ * L_ * H_ * 16;
    rope_kernel<<<nrope / 256, blk, 0, stream>>>(q, L_);
    rope_kernel<<<nrope / 256, blk, 0, stream>>>(k, L_);

    attn_kernel<<<gAt, blk, 0, stream>>>(q, k, v, ao);

    gemm_bf16_kernel<<<gN1, blk, 0, stream>>>(ao, wto, nullptr, tmp, M_, D_, HD_, 0);
    ln_kernel<<<M_, blk, 0, stream>>>(tmp, x, g1, be1, x, 1.0f);      // x <- x1

    gemm_bf16_kernel<<<gN4, blk, 0, stream>>>(x, wt1, c1, hbuf, M_, DFF_, D_, 1);
    gemm_bf16_kernel<<<gN1, blk, 0, stream>>>(hbuf, wt2, c2, tmp, M_, D_, DFF_, 0);

    float* dst = (layer == NL_ - 1) ? out : x;
    ln_kernel<<<M_, blk, 0, stream>>>(tmp, x, g2, be2, dst, 1.0f);    // x <- x2
  }
}
